// MambaBlock_85272280695014
// MI455X (gfx1250) — matrix operand order repaired and checked by reading, not compiled, not run on hardware
//
#include <hip/hip_runtime.h>
#include <hip/hip_bf16.h>
#include <math.h>

// ---- problem constants (from reference) ----
#define D_MODEL  768
#define D_STATE  16
#define D_CONV   4
#define D_INNER  1536
#define DT_RANK  48
#define NBATCH   2
#define SEQLEN   2048
#define NROWS    (NBATCH * SEQLEN)        // 4096
#define XPROJ_N  (DT_RANK + 2 * D_STATE)  // 80
#define XPROJ_NP 128                      // padded to BN
#define DTRK_P   64                       // DT_RANK padded to 2*TILE_K
#define LN_EPS   1e-5f

typedef __attribute__((ext_vector_type(16))) _Float16     v16h;
typedef __attribute__((ext_vector_type(8)))  float        v8f;
typedef __attribute__((ext_vector_type(4)))  unsigned int v4u;
typedef __attribute__((ext_vector_type(4)))  int          v4i;

// ---------------- WMMA GEMM (all-f16 operands, guard-free) ----------------
// C[M,N](f32) = act( A[M,K](f16, row stride lda) * W[N,K](f16, row stride K)^T + bias )
// Preconditions: M%64==0, N%128==0, K%32==0, lda%8==0 — enforced by padding.
#define TILE_M 64
#define TILE_N 128
#define TILE_K 32
#define LDSK   40   // padded f16 row stride: 80B -> bank-conflict-free, 16B aligned

#if defined(__has_builtin)
#if __has_builtin(__builtin_amdgcn_global_load_async_to_lds_b128)
#define HAVE_ASYNC_LDS 1
#endif
#endif

__device__ __forceinline__ float softplus_f(float x) {
  return (x > 20.f) ? x : log1pf(__expf(x));
}

__device__ __forceinline__ void copy16B_g2lds(const _Float16* g, _Float16* l) {
#if defined(HAVE_ASYNC_LDS)
  // signature: (int4 addrspace(1)*, int4 addrspace(3)*, imm offset, imm cpol)
  __builtin_amdgcn_global_load_async_to_lds_b128(
      (__attribute__((address_space(1))) v4i*)g,
      (__attribute__((address_space(3))) v4i*)l, 0, 0);
#else
  *(v4u*)l = *(const v4u*)g;
#endif
}

// wait until at most N async global->LDS ops remain outstanding (in-order completion)
__device__ __forceinline__ void wait_async_le3() {
#if defined(HAVE_ASYNC_LDS)
#if __has_builtin(__builtin_amdgcn_s_wait_asynccnt)
  __builtin_amdgcn_s_wait_asynccnt(3);
#else
  asm volatile("s_wait_asynccnt 3" ::: "memory");
#endif
#endif
}
__device__ __forceinline__ void wait_async_le0() {
#if defined(HAVE_ASYNC_LDS)
#if __has_builtin(__builtin_amdgcn_s_wait_asynccnt)
  __builtin_amdgcn_s_wait_asynccnt(0);
#else
  asm volatile("s_wait_asynccnt 0" ::: "memory");
#endif
#endif
}

// EPI: 0 = plain store, 1 = bias + softplus
template <int EPI>
__global__ __launch_bounds__(256)
void gemm_wmma_f16(float* __restrict__ C, const _Float16* __restrict__ A,
                   const _Float16* __restrict__ W, int M, int N, int K, int lda,
                   const float* __restrict__ bias)
{
  // double-buffered LDS tiles (A: 64x32, W: 128x32, f16, padded stride)
  __shared__ __align__(16) _Float16 As[2][TILE_M * LDSK];
  __shared__ __align__(16) _Float16 Ws[2][TILE_N * LDSK];

  const int t    = threadIdx.x;
  const int lane = t & 31;
  const int wave = t >> 5;
  const int half = lane >> 4;           // 0/1
  const int l16  = lane & 15;
  const int tm   = blockIdx.y * TILE_M;
  const int tn   = blockIdx.x * TILE_N;
  const int wm   = (wave >> 1) * 16;    // 0,16,32,48
  const int wn   = (wave & 1) * 64;     // 0 or 64

  // staging coordinates: one 16B (8 f16) chunk per thread per 64x32 half-tile
  const int ai  = t >> 2;               // 0..63
  const int aj  = (t & 3) * 8;          // 0,8,16,24
  const int wi0 = t >> 2;               // W rows 0..63
  const int wi1 = 64 + (t >> 2);        // W rows 64..127

  auto stage = [&](int kt, int buf) {
    const int k0 = kt * TILE_K;
    copy16B_g2lds(&A[(size_t)(tm + ai)  * lda + k0 + aj], &As[buf][ai  * LDSK + aj]);
    copy16B_g2lds(&W[(size_t)(tn + wi0) * K   + k0 + aj], &Ws[buf][wi0 * LDSK + aj]);
    copy16B_g2lds(&W[(size_t)(tn + wi1) * K   + k0 + aj], &Ws[buf][wi1 * LDSK + aj]);
  };

  union F8  { v8f  v; float    e[8];  };
  union F16 { v16h v; _Float16 e[16]; };
  F8 acc[4];
#pragma unroll
  for (int s = 0; s < 4; ++s)
#pragma unroll
    for (int i = 0; i < 8; ++i) acc[s].e[i] = 0.f;

  const int ktiles = K / TILE_K;
  stage(0, 0);                          // prologue: tile 0 -> buffer 0

  for (int kt = 0; kt < ktiles; ++kt) {
    const int buf = kt & 1;
    const bool more = (kt + 1 < ktiles);
    if (more) {
      stage(kt + 1, buf ^ 1);           // overlap next tile's loads with compute
      wait_async_le3();                 // current tile (older 3 loads) has landed
    } else {
      wait_async_le0();
    }
    __syncthreads();

    // A fragment: m = wm + l16; e[2v+s] holds k = (v/4)*16 + half*8 + (v%4)*2 + s
    F16 af;
    const _Float16* arow = &As[buf][(wm + l16) * LDSK];
#pragma unroll
    for (int g = 0; g < 2; ++g)
#pragma unroll
      for (int j = 0; j < 8; ++j)
        af.e[g * 8 + j] = arow[g * 16 + half * 8 + j];

    // 4 subtiles of 16 cols each: B e[2v+s] holds k = (v/4)*16 + half*8 + (v%4)*2 + s
#pragma unroll
    for (int s = 0; s < 4; ++s) {
      F16 bf;
      const _Float16* wrow = &Ws[buf][(wn + s * 16 + l16) * LDSK];
#pragma unroll
      for (int g = 0; g < 2; ++g)
#pragma unroll
        for (int j = 0; j < 8; ++j) bf.e[g * 8 + j] = wrow[g * 16 + half * 8 + j];
      acc[s].v = __builtin_amdgcn_wmma_f32_16x16x32_f16(
          false, af.v, false, bf.v, (short)0, acc[s].v, false, false);
    }
    __syncthreads();
  }

  // epilogue: C/D layout -> m = r + half*8, n = l16
#pragma unroll
  for (int s = 0; s < 4; ++s) {
#pragma unroll
    for (int r = 0; r < 8; ++r) {
      const int gm = tm + wm + half * 8 + r;
      const int gn = tn + wn + s * 16 + l16;
      float v = acc[s].e[r];
      if (EPI == 1) v = softplus_f(v + bias[gn]);
      C[(size_t)gm * N + gn] = v;
    }
  }
}

// -------- fp32 -> f16 copy with zero padding: dst[RPxCP] from src[RxC] --------
__global__ __launch_bounds__(256)
void cvt_pad_f16(const float* __restrict__ src, _Float16* __restrict__ dst,
                 int R, int Cc, int RP, int CP)
{
  const int idx = blockIdx.x * blockDim.x + threadIdx.x;
  if (idx >= RP * CP) return;
  const int r = idx / CP, c = idx % CP;
  const float v = (r < R && c < Cc) ? src[(size_t)r * Cc + c] : 0.f;
  dst[idx] = (_Float16)v;
}

// -------- pack dtl (x_dbl cols 0..47) into f16 [NROWS x 64], zero pad --------
__global__ __launch_bounds__(256)
void pack_dtl_kernel(const float* __restrict__ xdbl, _Float16* __restrict__ dtl16)
{
  const int idx = blockIdx.x * blockDim.x + threadIdx.x;  // NROWS*64
  const int r = idx >> 6, c = idx & 63;
  const float v = (c < DT_RANK) ? xdbl[(size_t)r * XPROJ_NP + c] : 0.f;
  dtl16[idx] = (_Float16)v;
}

// ---------------- depthwise causal conv (k=4) + SiLU ----------------
__global__ __launch_bounds__(256)
void conv_silu_kernel(const float* __restrict__ xz, const float* __restrict__ cw,
                      const float* __restrict__ cb, float* __restrict__ u,
                      _Float16* __restrict__ u16, int total)
{
  const int idx = blockIdx.x * blockDim.x + threadIdx.x;
  if (idx >= total) return;
  const int d  = idx % D_INNER;
  const int bt = idx / D_INNER;       // b*L + t
  const int tt = bt % SEQLEN;
  float acc = cb[d];
#pragma unroll
  for (int j = 0; j < D_CONV; ++j) {
    const int src = tt - (D_CONV - 1) + j;
    if (src >= 0)
      acc += cw[d * D_CONV + j] * xz[(size_t)(bt - (D_CONV - 1) + j) * (2 * D_INNER) + d];
  }
  const float s = 1.f / (1.f + __expf(-acc));
  const float uv = acc * s;
  u[idx]   = uv;
  u16[idx] = (_Float16)uv;
}

// ---------------- selective scan (sequential over L) ----------------
__global__ __launch_bounds__(256)
void scan_kernel(const float* __restrict__ x_dbl, const float* __restrict__ dt,
                 const float* __restrict__ u, const float* __restrict__ xz,
                 const float* __restrict__ A_log, const float* __restrict__ Dv,
                 _Float16* __restrict__ yg16)
{
  const int gid = blockIdx.x * blockDim.x + threadIdx.x;  // 0..3071
  const int b = gid / D_INNER;
  const int d = gid % D_INNER;

  float Arow[D_STATE];
#pragma unroll
  for (int n = 0; n < D_STATE; ++n) Arow[n] = -__expf(A_log[d * D_STATE + n]);
  float h[D_STATE];
#pragma unroll
  for (int n = 0; n < D_STATE; ++n) h[n] = 0.f;
  const float Dd = Dv[d];

  for (int tt = 0; tt < SEQLEN; ++tt) {
    const size_t row = (size_t)b * SEQLEN + tt;
    const float* xr = x_dbl + row * XPROJ_NP;       // [48 dt | 16 B | 16 C | pad]
    const float dtv = dt[row * D_INNER + d];
    const float uv  = u [row * D_INNER + d];
    const float zv  = xz[row * (2 * D_INNER) + D_INNER + d];
    const float du  = dtv * uv;
    float y = 0.f;
#pragma unroll
    for (int n = 0; n < D_STATE; ++n) {
      const float dA = __expf(dtv * Arow[n]);
      h[n] = h[n] * dA + du * xr[DT_RANK + n];
      y += h[n] * xr[DT_RANK + D_STATE + n];
    }
    y += uv * Dd;
    const float sg = 1.f / (1.f + __expf(-zv));
    yg16[row * D_INNER + d] = (_Float16)(y * (zv * sg));
  }
}

// ---------------- residual + LayerNorm ----------------
__global__ __launch_bounds__(256)
void ln_kernel(const float* __restrict__ o, const float* __restrict__ res,
               const float* __restrict__ w, const float* __restrict__ bia,
               float* __restrict__ out)
{
  __shared__ float s1[256], s2[256];
  const int row = blockIdx.x;
  const float* op = o   + (size_t)row * D_MODEL;
  const float* rp = res + (size_t)row * D_MODEL;
  float vals[3], lsum = 0.f, lsq = 0.f;
#pragma unroll
  for (int i = 0; i < 3; ++i) {
    const float hv = op[threadIdx.x + i * 256] + rp[threadIdx.x + i * 256];
    vals[i] = hv; lsum += hv; lsq += hv * hv;
  }
  s1[threadIdx.x] = lsum; s2[threadIdx.x] = lsq;
  __syncthreads();
  for (int off = 128; off > 0; off >>= 1) {
    if (threadIdx.x < off) {
      s1[threadIdx.x] += s1[threadIdx.x + off];
      s2[threadIdx.x] += s2[threadIdx.x + off];
    }
    __syncthreads();
  }
  const float mu   = s1[0] * (1.f / D_MODEL);
  const float var  = s2[0] * (1.f / D_MODEL) - mu * mu;
  const float rstd = rsqrtf(var + LN_EPS);
#pragma unroll
  for (int i = 0; i < 3; ++i) {
    const int c = threadIdx.x + i * 256;
    out[(size_t)row * D_MODEL + c] = (vals[i] - mu) * rstd * w[c] + bia[c];
  }
}

// ---------------- launch ----------------
static inline size_t align256(size_t x) { return (x + 255) & ~(size_t)255; }

extern "C" void kernel_launch(void* const* d_in, const int* in_sizes, int n_in,
                              void* d_out, int out_size, void* d_ws, size_t ws_size,
                              hipStream_t stream) {
  const float* x          = (const float*)d_in[0];   // [2,2048,768]
  const float* in_proj_w  = (const float*)d_in[1];   // [3072,768]
  const float* conv_w     = (const float*)d_in[2];   // [1536,1,4]
  const float* conv_b     = (const float*)d_in[3];   // [1536]
  const float* x_proj_w   = (const float*)d_in[4];   // [80,1536]
  const float* dt_proj_w  = (const float*)d_in[5];   // [1536,48]
  const float* dt_proj_b  = (const float*)d_in[6];   // [1536]
  const float* A_log      = (const float*)d_in[7];   // [1536,16]
  const float* Dv         = (const float*)d_in[8];   // [1536]
  const float* out_proj_w = (const float*)d_in[9];   // [768,1536]
  const float* ln_w       = (const float*)d_in[10];  // [768]
  const float* ln_b       = (const float*)d_in[11];  // [768]
  float* outp = (float*)d_out;                       // [2,2048,768]

  // ---- workspace carve (all 256B aligned) ----
  char* p = (char*)d_ws;
  auto carve = [&](size_t bytes) { char* r = p; p += align256(bytes); return r; };
  float*    xz     = (float*)   carve((size_t)NROWS * 2 * D_INNER * 4);   // 50.3MB
  float*    u      = (float*)   carve((size_t)NROWS * D_INNER * 4);       // 25.2MB
  float*    xdbl   = (float*)   carve((size_t)NROWS * XPROJ_NP * 4);      //  2.1MB
  float*    dt     = (float*)   carve((size_t)NROWS * D_INNER * 4);       // 25.2MB
  float*    obuf   = (float*)   carve((size_t)NROWS * D_MODEL * 4);       // 12.6MB
  _Float16* x16    = (_Float16*)carve((size_t)NROWS * D_MODEL * 2);       //  6.3MB
  _Float16* u16    = (_Float16*)carve((size_t)NROWS * D_INNER * 2);       // 12.6MB
  _Float16* dtl16  = (_Float16*)carve((size_t)NROWS * DTRK_P * 2);        //  0.5MB
  _Float16* yg16   = (_Float16*)carve((size_t)NROWS * D_INNER * 2);       // 12.6MB
  _Float16* w_in16 = (_Float16*)carve((size_t)2 * D_INNER * D_MODEL * 2); //  4.7MB
  _Float16* w_xp16 = (_Float16*)carve((size_t)XPROJ_NP * D_INNER * 2);    //  0.4MB
  _Float16* w_dt16 = (_Float16*)carve((size_t)D_INNER * DTRK_P * 2);      //  0.2MB
  _Float16* w_ot16 = (_Float16*)carve((size_t)D_MODEL * D_INNER * 2);     //  2.4MB

  const dim3 blk(256);
  auto blocks = [](size_t n) { return dim3((unsigned)((n + 255) / 256)); };

  // ---- 0) f16 conversions / zero-padding ----
  cvt_pad_f16<<<blocks((size_t)NROWS * D_MODEL), blk, 0, stream>>>(
      x, x16, NROWS, D_MODEL, NROWS, D_MODEL);
  cvt_pad_f16<<<blocks((size_t)2 * D_INNER * D_MODEL), blk, 0, stream>>>(
      in_proj_w, w_in16, 2 * D_INNER, D_MODEL, 2 * D_INNER, D_MODEL);
  cvt_pad_f16<<<blocks((size_t)XPROJ_NP * D_INNER), blk, 0, stream>>>(
      x_proj_w, w_xp16, XPROJ_N, D_INNER, XPROJ_NP, D_INNER);
  cvt_pad_f16<<<blocks((size_t)D_INNER * DTRK_P), blk, 0, stream>>>(
      dt_proj_w, w_dt16, D_INNER, DT_RANK, D_INNER, DTRK_P);
  cvt_pad_f16<<<blocks((size_t)D_MODEL * D_INNER), blk, 0, stream>>>(
      out_proj_w, w_ot16, D_MODEL, D_INNER, D_MODEL, D_INNER);

  // ---- 1) in_proj: xz = x @ in_proj_w^T   [4096 x 3072], K=768 ----
  gemm_wmma_f16<0><<<dim3((2 * D_INNER) / TILE_N, NROWS / TILE_M), blk, 0, stream>>>(
      xz, x16, w_in16, NROWS, 2 * D_INNER, D_MODEL, D_MODEL, nullptr);

  // ---- 2) causal depthwise conv + SiLU -> u (f32 + f16) ----
  conv_silu_kernel<<<blocks((size_t)NROWS * D_INNER), blk, 0, stream>>>(
      xz, conv_w, conv_b, u, u16, NROWS * D_INNER);

  // ---- 3) x_proj: x_dbl = u @ x_proj_w^T   [4096 x 128(pad)], K=1536 ----
  gemm_wmma_f16<0><<<dim3(XPROJ_NP / TILE_N, NROWS / TILE_M), blk, 0, stream>>>(
      xdbl, u16, w_xp16, NROWS, XPROJ_NP, D_INNER, D_INNER, nullptr);

  // ---- 3b) pack dtl -> f16 [4096 x 64] (zeros beyond 48) ----
  pack_dtl_kernel<<<blocks((size_t)NROWS * DTRK_P), blk, 0, stream>>>(xdbl, dtl16);

  // ---- 4) dt = softplus(dtl @ dt_proj_w^T + b)   [4096 x 1536], K=64(pad) ----
  gemm_wmma_f16<1><<<dim3(D_INNER / TILE_N, NROWS / TILE_M), blk, 0, stream>>>(
      dt, dtl16, w_dt16, NROWS, D_INNER, DTRK_P, DTRK_P, dt_proj_b);

  // ---- 5) selective scan + D skip + SiLU(z) gate -> yg16 ----
  scan_kernel<<<(NBATCH * D_INNER) / 256, blk, 0, stream>>>(
      xdbl, dt, u, xz, A_log, Dv, yg16);

  // ---- 6) out_proj: obuf = yg @ out_proj_w^T   [4096 x 768], K=1536 ----
  gemm_wmma_f16<0><<<dim3(D_MODEL / TILE_N, NROWS / TILE_M), blk, 0, stream>>>(
      obuf, yg16, w_ot16, NROWS, D_MODEL, D_INNER, D_INNER, nullptr);

  // ---- 7) residual + LayerNorm -> d_out ----
  ln_kernel<<<NROWS, blk, 0, stream>>>(obuf, x, ln_w, ln_b, outp);
}
